// SparseAttention_38585986187456
// MI455X (gfx1250) — matrix operand order repaired and checked by reading, not compiled, not run on hardware
//
#include <hip/hip_runtime.h>
#include <hip/hip_bf16.h>
#include <stdint.h>

typedef __attribute__((ext_vector_type(4)))  _Float16 v4h;
typedef __attribute__((ext_vector_type(8)))  _Float16 v8h;
typedef __attribute__((ext_vector_type(16))) _Float16 v16h;
typedef __attribute__((ext_vector_type(8)))  float    v8f;

#define EMBED 1024
#define SEQ   2048
#define BATCH 4
#define MROWS (BATCH * SEQ)   // 8192
#define TOPK  32

// ---------------------------------------------------------------------------
// CDNA5 async global->LDS copy (one b128 per lane), tracked by ASYNCcnt.
// ---------------------------------------------------------------------------
__device__ __forceinline__ void async_b128(uint32_t lds_off, uint32_t voff, uint64_t sbase) {
    asm volatile("global_load_async_to_lds_b128 %0, %1, %2"
                 :: "v"(lds_off), "v"(voff), "s"(sbase) : "memory");
}

__device__ __forceinline__ void wait_async0() {
#if __has_builtin(__builtin_amdgcn_s_wait_asynccnt)
    __builtin_amdgcn_s_wait_asynccnt(0);
#else
    asm volatile("s_wait_asynccnt 0x0" ::: "memory");
#endif
}

// A fragment (ISA 16-bit 16x32 A layout): two 8-half runs per lane.
__device__ __forceinline__ v16h load_afrag(const _Float16* __restrict__ arow, int k0, int grp) {
    v8h lo = *(const v8h*)(arow + k0 + 8 * grp);
    v8h hi = *(const v8h*)(arow + k0 + 16 + 8 * grp);
    return __builtin_shufflevector(lo, hi, 0,1,2,3,4,5,6,7,8,9,10,11,12,13,14,15);
}

// ---------------------------------------------------------------------------
// f32 -> f16 conversion (vectorized 4-wide)
// ---------------------------------------------------------------------------
__global__ __launch_bounds__(256)
void f32_to_f16_kernel(const float* __restrict__ src, _Float16* __restrict__ dst, int n4) {
    int i = blockIdx.x * blockDim.x + threadIdx.x;
    if (i >= n4) return;
    float4 v = ((const float4*)src)[i];
    v4h h;
    h[0] = (_Float16)v.x; h[1] = (_Float16)v.y;
    h[2] = (_Float16)v.z; h[3] = (_Float16)v.w;
    ((v4h*)dst)[i] = h;
}

// ---------------------------------------------------------------------------
// NT WMMA GEMM: C[m,n] = sum_k A[m,k]*B[n,k] + bias[n]
// Block tile: 128 rows x 128 cols (8 waves x 16-row stripes, 8 acc each).
// 128x32 B slice shared by all waves via async-to-LDS double buffering
// (2 async b128 per thread per slice). All 8 B fragments are read from LDS
// before the 8 WMMAs -> one dscnt wait per 8 matrix ops. A fragments are
// register double-buffered. __launch_bounds__(256,1) releases the register
// ceiling so the batched fragments stay resident (acc 64 + bf 64 + af 16).
// grid.x = (M/128)*(N/128), blockDim = 256.
// ---------------------------------------------------------------------------
__global__ __launch_bounds__(256, 1)
void gemm_nt_wmma(const _Float16* __restrict__ A, const _Float16* __restrict__ Bm,
                  const float* __restrict__ bias, void* __restrict__ Cout,
                  int M, int N, int K, int c_f16) {
    __shared__ __align__(16) _Float16 bstage[2][128 * 32];   // 2 x 8KB

    int tid  = threadIdx.x;
    int wave = tid >> 5;
    int lane = tid & 31;
    int grp  = lane >> 4;
    int l16  = lane & 15;

    int nt8 = N >> 7;
    int mb  = blockIdx.x / nt8;       // 128-row block
    int nt  = blockIdx.x % nt8;       // 128-col group
    int mt  = mb * 8 + wave;          // this wave's 16-row tile

    const _Float16* arow = A + (size_t)(mt * 16 + l16) * K;

    // staging: 128 rows x 64B = 512 chunks of 16B; 2 chunks per thread
    int c0 = tid, c1 = tid + 256;
    int n0 = c0 >> 2, ch0 = c0 & 3;
    int n1 = c1 >> 2, ch1 = c1 & 3;
    uint64_t sbase  = (uint64_t)(uintptr_t)(Bm + (size_t)nt * 128 * K);
    uint32_t voffA  = (uint32_t)((n0 * K + ch0 * 8) * 2);
    uint32_t voffB  = (uint32_t)((n1 * K + ch1 * 8) * 2);
    uint32_t ldsbase = (uint32_t)(uintptr_t)(&bstage[0][0]);
    uint32_t ldsA   = ldsbase + (uint32_t)(n0 * 32 + ch0 * 8) * 2;
    uint32_t ldsB   = ldsbase + (uint32_t)(n1 * 32 + ch1 * 8) * 2;

    v8f acc[8];
#pragma unroll
    for (int s = 0; s < 8; ++s) {
        float bv = bias[nt * 128 + s * 16 + l16];
#pragma unroll
        for (int r = 0; r < 8; ++r) acc[s][r] = bv;
    }

    // prologue: stage k-slice 0 into buffer 0, prefetch first A fragment
    async_b128(ldsA, voffA, sbase);
    async_b128(ldsB, voffB, sbase);
    v16h af_cur = load_afrag(arow, 0, grp);

    int cur = 0;
    for (int k0 = 0; k0 < K; k0 += 32) {
        wait_async0();       // own async chunks for buf[cur] complete
        __syncthreads();     // all waves' chunks visible; prev readers done
        if (k0 + 32 < K) {   // prefetch next slice into alternate buffer
            uint32_t bufoff = (uint32_t)((cur ^ 1) * 128 * 32 * 2);
            uint32_t koff   = (uint32_t)((k0 + 32) * 2);
            async_b128(ldsA + bufoff, voffA + koff, sbase);
            async_b128(ldsB + bufoff, voffB + koff, sbase);
        }
        v16h af_nxt = af_cur;
        if (k0 + 32 < K) af_nxt = load_afrag(arow, k0 + 32, grp);

        // batch all 8 B-fragment LDS reads, then fire 8 WMMAs back-to-back
        const _Float16* bufp = &bstage[cur][0];
        v16h bf[8];
#pragma unroll
        for (int s = 0; s < 8; ++s) {
            const _Float16* bp = bufp + ((s * 16 + l16) * 32 + 8 * grp);
            v8h blo = *(const v8h*)bp;
            v8h bhi = *(const v8h*)(bp + 16);
            bf[s] = __builtin_shufflevector(blo, bhi,
                0,1,2,3,4,5,6,7,8,9,10,11,12,13,14,15);
        }
#pragma unroll
        for (int s = 0; s < 8; ++s)
            acc[s] = __builtin_amdgcn_wmma_f32_16x16x32_f16(
                false, af_cur, false, bf[s], (short)0, acc[s], false, false);

        af_cur = af_nxt;
        cur ^= 1;
    }

#pragma unroll
    for (int s = 0; s < 8; ++s) {
        int col = nt * 128 + s * 16 + l16;
#pragma unroll
        for (int r = 0; r < 8; ++r) {
            int rowm = mt * 16 + r + 8 * grp;
            if (c_f16) ((_Float16*)Cout)[(size_t)rowm * N + col] = (_Float16)acc[s][r];
            else       ((float*)Cout)[(size_t)rowm * N + col]    = acc[s][r];
        }
    }
}

// ---------------------------------------------------------------------------
// Fused: scores (Q*K^T/32 via WMMA, 16x2048 f32 stripe resident in LDS) ->
// exact top-32 per row -> softmax -> sparse attn*V gather -> f16 context.
// grid.x = BATCH*(SEQ/16) = 512 blocks of 256 threads (8 waves).
// ---------------------------------------------------------------------------
#define KT_PER_WAVE 16   // 128 k-tiles / 8 waves

__device__ __forceinline__ v16h load_kfrag(const _Float16* __restrict__ Kb,
                                           int kt, int l16, int e0, int grp) {
    const _Float16* kr = Kb + (size_t)(kt * 16 + l16) * EMBED + e0 + 8 * grp;
    v8h lo = *(const v8h*)(kr);
    v8h hi = *(const v8h*)(kr + 16);
    return __builtin_shufflevector(lo, hi, 0,1,2,3,4,5,6,7,8,9,10,11,12,13,14,15);
}

__global__ __launch_bounds__(256, 1)
void sparse_attn_fused(const _Float16* __restrict__ Q,
                       const _Float16* __restrict__ Km,
                       const _Float16* __restrict__ V,
                       _Float16* __restrict__ O) {
    extern __shared__ char smem[];
    _Float16* qs   = (_Float16*)smem;                               // 16x1024 f16
    float*    sc   = (float*)(smem + 16 * EMBED * 2);               // 16x2048 f32
    float*    topv = (float*)(smem + 16 * EMBED * 2 + 16 * SEQ * 4);
    int*      topi = (int*)((char*)topv + 16 * TOPK * 4);
    float*    wts  = (float*)((char*)topi + 16 * TOPK * 4);

    int tid  = threadIdx.x;
    int wave = tid >> 5;
    int lane = tid & 31;
    int grp  = lane >> 4;
    int l16  = lane & 15;

    int b  = blockIdx.x >> 7;
    int qt = blockIdx.x & 127;
    size_t qbase = ((size_t)b * SEQ + (size_t)qt * 16) * EMBED;

    // ---- stage 16x1024 Q tile into LDS via async-to-LDS (32KB, 8 x b128/lane)
    {
        uint32_t qs_off = (uint32_t)(uintptr_t)qs;
        uint64_t qsrc   = (uint64_t)(uintptr_t)(Q + qbase);
#pragma unroll
        for (int j = 0; j < 8; ++j) {
            uint32_t c = (uint32_t)(tid + 256 * j) * 16;  // byte offset
            async_b128(qs_off + c, c, qsrc);
        }
        wait_async0();
        __syncthreads();
    }

    // ---- scores: each wave owns 16 k-tiles; A frag reused 16x; B paired
    v8f acc[KT_PER_WAVE];
#pragma unroll
    for (int t = 0; t < KT_PER_WAVE; ++t)
#pragma unroll
        for (int r = 0; r < 8; ++r) acc[t][r] = 0.0f;

    const _Float16* Kb = Km + (size_t)b * SEQ * EMBED;
    int kt0 = wave * KT_PER_WAVE;
    for (int e0 = 0; e0 < EMBED; e0 += 32) {
        const _Float16* qr = qs + (size_t)l16 * EMBED;
        v8h alo = *(const v8h*)(qr + e0 + 8 * grp);
        v8h ahi = *(const v8h*)(qr + e0 + 16 + 8 * grp);
        v16h af = __builtin_shufflevector(alo, ahi,
            0,1,2,3,4,5,6,7,8,9,10,11,12,13,14,15);

        v16h bf0 = load_kfrag(Kb, kt0 + 0, l16, e0, grp);
        v16h bf1 = load_kfrag(Kb, kt0 + 1, l16, e0, grp);
#pragma unroll
        for (int t = 0; t < KT_PER_WAVE; t += 2) {
            v16h n0 = bf0, n1 = bf1;
            if (t + 2 < KT_PER_WAVE) {   // prefetch next pair
                n0 = load_kfrag(Kb, kt0 + t + 2, l16, e0, grp);
                n1 = load_kfrag(Kb, kt0 + t + 3, l16, e0, grp);
            }
            acc[t]     = __builtin_amdgcn_wmma_f32_16x16x32_f16(
                false, af, false, bf0, (short)0, acc[t], false, false);
            acc[t + 1] = __builtin_amdgcn_wmma_f32_16x16x32_f16(
                false, af, false, bf1, (short)0, acc[t + 1], false, false);
            bf0 = n0; bf1 = n1;
        }
    }
    const float inv_scale = 0.03125f;   // 1/sqrt(1024)
#pragma unroll
    for (int t = 0; t < KT_PER_WAVE; ++t) {
        int kt = kt0 + t;
#pragma unroll
        for (int r = 0; r < 8; ++r)
            sc[(size_t)(r + 8 * grp) * SEQ + kt * 16 + l16] = acc[t][r] * inv_scale;
    }
    __syncthreads();

    // ---- exact top-32 per row (2 rows per wave, iterative arg-max) ----
    for (int rr = 0; rr < 2; ++rr) {
        int row = wave * 2 + rr;
        float* srow = sc + (size_t)row * SEQ;
        for (int i = 0; i < TOPK; ++i) {
            float bv = -1e30f; int bi = 0;
            for (int j = lane; j < SEQ; j += 32) {
                float v = srow[j];
                if (v > bv) { bv = v; bi = j; }
            }
#pragma unroll
            for (int off = 16; off > 0; off >>= 1) {
                float ov = __shfl_xor(bv, off);
                int   oi = __shfl_xor(bi, off);
                if (ov > bv || (ov == bv && oi < bi)) { bv = ov; bi = oi; }
            }
            if (lane == 0) {
                topv[row * TOPK + i] = bv;
                topi[row * TOPK + i] = bi;
                srow[bi] = -1e30f;   // same-wave LDS ops are in-order
            }
        }
        float v = topv[row * TOPK + lane];
        float m = topv[row * TOPK + 0];
        float e = __expf(v - m);
        float s = e;
#pragma unroll
        for (int off = 16; off > 0; off >>= 1) s += __shfl_xor(s, off);
        wts[row * TOPK + lane] = e / s;
    }
    __syncthreads();

    // ---- sparse gather: out[r, c] = sum_i w_i * V[idx_i, c] ----
    const _Float16* Vb = V + (size_t)b * SEQ * EMBED;
    int c0 = tid * 4;
    for (int r = 0; r < 16; ++r) {
        float a0 = 0.f, a1 = 0.f, a2 = 0.f, a3 = 0.f;
#pragma unroll
        for (int i = 0; i < TOPK; ++i) {
            float w  = wts[r * TOPK + i];
            int  idx = topi[r * TOPK + i];
            v4h vv = *(const v4h*)(Vb + (size_t)idx * EMBED + c0);
            a0 += w * (float)vv[0]; a1 += w * (float)vv[1];
            a2 += w * (float)vv[2]; a3 += w * (float)vv[3];
        }
        v4h ov;
        ov[0] = (_Float16)a0; ov[1] = (_Float16)a1;
        ov[2] = (_Float16)a2; ov[3] = (_Float16)a3;
        *(v4h*)(O + qbase + (size_t)r * EMBED + c0) = ov;
    }
}

// ---------------------------------------------------------------------------
// Host-side orchestration
// ---------------------------------------------------------------------------
extern "C" void kernel_launch(void* const* d_in, const int* in_sizes, int n_in,
                              void* d_out, int out_size, void* d_ws, size_t ws_size,
                              hipStream_t stream) {
    const float* x  = (const float*)d_in[0];
    const float* Wq = (const float*)d_in[1];
    const float* bq = (const float*)d_in[2];
    const float* Wk = (const float*)d_in[3];
    const float* bk = (const float*)d_in[4];
    const float* Wv = (const float*)d_in[5];
    const float* bv = (const float*)d_in[6];
    const float* Wo = (const float*)d_in[7];
    const float* bo = (const float*)d_in[8];

    char* ws = (char*)d_ws;
    const size_t XB = (size_t)MROWS * EMBED * 2;   // 16 MB f16
    const size_t WB = (size_t)EMBED * EMBED * 2;   //  2 MB f16
    _Float16* x16  = (_Float16*)(ws);
    _Float16* Wq16 = (_Float16*)(ws + XB);
    _Float16* Wk16 = (_Float16*)(ws + XB + WB);
    _Float16* Wv16 = (_Float16*)(ws + XB + 2 * WB);
    _Float16* Wo16 = (_Float16*)(ws + XB + 3 * WB);
    _Float16* Q16  = (_Float16*)(ws + XB + 4 * WB);
    _Float16* K16  = (_Float16*)(ws + 2 * XB + 4 * WB);
    _Float16* V16  = (_Float16*)(ws + 3 * XB + 4 * WB);
    _Float16* O16  = (_Float16*)(ws + 4 * XB + 4 * WB);

    {
        int n4 = MROWS * EMBED / 4;
        f32_to_f16_kernel<<<n4 / 256, 256, 0, stream>>>(x, x16, n4);
        int w4 = EMBED * EMBED / 4;
        f32_to_f16_kernel<<<w4 / 256, 256, 0, stream>>>(Wq, Wq16, w4);
        f32_to_f16_kernel<<<w4 / 256, 256, 0, stream>>>(Wk, Wk16, w4);
        f32_to_f16_kernel<<<w4 / 256, 256, 0, stream>>>(Wv, Wv16, w4);
        f32_to_f16_kernel<<<w4 / 256, 256, 0, stream>>>(Wo, Wo16, w4);
    }

    // QKV projections: 8192x1024 @ (1024x1024)^T, f16 out
    int gemm_blocks = (MROWS / 128) * (EMBED / 128);   // 64*8 = 512
    gemm_nt_wmma<<<gemm_blocks, 256, 0, stream>>>(x16, Wq16, bq, Q16, MROWS, EMBED, EMBED, 1);
    gemm_nt_wmma<<<gemm_blocks, 256, 0, stream>>>(x16, Wk16, bk, K16, MROWS, EMBED, EMBED, 1);
    gemm_nt_wmma<<<gemm_blocks, 256, 0, stream>>>(x16, Wv16, bv, V16, MROWS, EMBED, EMBED, 1);

    // fused scores/top-k/softmax/gather: 170KB LDS (gfx1250 WGP has 320KB)
    size_t smem = (size_t)16 * EMBED * 2 + (size_t)16 * SEQ * 4 + 3 * 16 * TOPK * 4;
    sparse_attn_fused<<<BATCH * (SEQ / 16), 256, smem, stream>>>(Q16, K16, V16, O16);

    // output projection, f32 out
    gemm_nt_wmma<<<gemm_blocks, 256, 0, stream>>>(O16, Wo16, bo, (float*)d_out,
                                                  MROWS, EMBED, EMBED, 0);
}
